// Qwen3NextMemGatedDeltaNet_10325101379651
// MI455X (gfx1250) — matrix operand order repaired and checked by reading, not compiled, not run on hardware
//
#include <hip/hip_runtime.h>

// ---------------------------------------------------------------------------
// Qwen3-Next gated delta net for MI455X (gfx1250, wave32, WMMA).
// B=2 S=2048 H=2048 NK=16 NV=32 DK=DV=128 KCONV=4 CHUNK=64
// ---------------------------------------------------------------------------

typedef __attribute__((ext_vector_type(2)))  float  v2f;
typedef __attribute__((ext_vector_type(8)))  float  v8f;
typedef __attribute__((ext_vector_type(16))) __bf16 v16bf;
typedef __attribute__((ext_vector_type(4)))  int    v4i_;

#define BQ 2
#define SQ 2048
#define BS 4096            // B*S
#define HQ 2048
#define NKH 16
#define NVH 32
#define DKQ 128
#define DVQ 128
#define QKVZ_N 12288       // 2*KD + 2*VD
#define CONVD 8192         // 2*KD + VD
#define HEADCOLS 768       // per-NK-head column stride in qkvz

// padded LDS strides (stride mod 64 banks == 4 -> conflict-free frag reads)
#define LDA_BF 40          // bf16 elems (80B, 16B aligned frag reads)
#define LDB_BF 40
#define LD128 132          // f32 dwords
#define LD64  68

#if __has_builtin(__builtin_amdgcn_global_load_async_to_lds_b128)
#define ASYNC_LDS 1
typedef __attribute__((address_space(1))) v4i_* gp4_t;
typedef __attribute__((address_space(3))) v4i_* lp4_t;
#else
#define ASYNC_LDS 0
#endif

__device__ __forceinline__ __bf16 f2bf(float f) {
  unsigned u = __float_as_uint(f);
  unsigned r = (u + 0x7fffu + ((u >> 16) & 1u)) >> 16;
  unsigned short s = (unsigned short)r;
  return __builtin_bit_cast(__bf16, s);
}
__device__ __forceinline__ unsigned pack2(__bf16 lo, __bf16 hi) {
  return (unsigned)__builtin_bit_cast(unsigned short, lo) |
         ((unsigned)__builtin_bit_cast(unsigned short, hi) << 16);
}

__device__ __forceinline__ float sigmoidf_(float x) { return 1.f / (1.f + __expf(-x)); }
__device__ __forceinline__ float siluf_(float x)    { return x * sigmoidf_(x); }

// ---------------------------------------------------------------------------
// fp32 -> bf16 cast
// ---------------------------------------------------------------------------
__global__ void cast_bf16_kernel(const float* __restrict__ src, __bf16* __restrict__ dst, int n) {
  int i = blockIdx.x * 256 + threadIdx.x;
  if (i < n) dst[i] = f2bf(src[i]);
}

// ---------------------------------------------------------------------------
// bf16 GEMM, f32 accumulate: C[M,N] = A[M,K] @ B[K,N]
// block tile 128x128, 256 threads = 8 waves (4x2), wave tile 32x64,
// K-step 32 via v_wmma_f32_16x16x32_bf16.
// A staged row-major (frag is K-contiguous, same k order as B frags);
// B staged TRANSPOSED (sBt[n][k]) so B frags are K-contiguous b128 loads.
// A tile staged with CDNA5 async global->LDS (ASYNCcnt) when available.
// ---------------------------------------------------------------------------
__global__ __launch_bounds__(256) void gemm_bf16_kernel(
    const __bf16* __restrict__ A, const __bf16* __restrict__ B,
    float* __restrict__ C, int M, int N, int K) {
  __shared__ __bf16 sA [128 * LDA_BF];   // [m][k]
  __shared__ __bf16 sBt[128 * LDB_BF];   // [n][k]

  const int tid  = threadIdx.x;
  const int lane = tid & 31;
  const int w    = tid >> 5;
  const int mblk = blockIdx.y * 128;
  const int nblk = blockIdx.x * 128;
  const int wm   = (w >> 1) * 32;   // 0,32,64,96
  const int wn   = (w & 1) * 64;    // 0,64
  const bool nfull = (nblk + 128) <= N;

  v8f acc[2][4];
#pragma unroll
  for (int i = 0; i < 2; ++i)
#pragma unroll
    for (int j = 0; j < 4; ++j)
      acc[i][j] = (v8f){0.f,0.f,0.f,0.f,0.f,0.f,0.f,0.f};

  for (int k0 = 0; k0 < K; k0 += 32) {
    // ---- stage A tile 128x32 (row-major) ----
    {
      int r = tid >> 1, cs = (tid & 1) * 16;
      const __bf16* src = A + (size_t)(mblk + r) * K + k0 + cs;
#if ASYNC_LDS
      __builtin_amdgcn_global_load_async_to_lds_b128(
          (gp4_t)(void*)src, (lp4_t)&sA[r * LDA_BF + cs], 0, 0);
#else
      *(uint4*)&sA[r * LDA_BF + cs] = *(const uint4*)src;
#endif
      if (k0 + 32 < K) __builtin_prefetch(src + 32, 0, 1);   // global_prefetch_b8
    }
    // ---- stage B tile 32x128, transposed into sBt[n][k] ----
    {
      int a2 = tid >> 4;            // row pair {2a2, 2a2+1}
      int cs = (tid & 15) * 8;      // 8 columns
      if (nfull) {
        const __bf16* s0 = B + (size_t)(k0 + 2 * a2) * N + nblk + cs;
        uint4 r0 = *(const uint4*)s0;
        uint4 r1 = *(const uint4*)(s0 + N);
        const __bf16* p0 = (const __bf16*)&r0;
        const __bf16* p1 = (const __bf16*)&r1;
#pragma unroll
        for (int e = 0; e < 8; ++e)
          *(unsigned*)&sBt[(cs + e) * LDB_BF + 2 * a2] = pack2(p0[e], p1[e]);
        if (k0 + 32 < K) __builtin_prefetch(s0 + (size_t)32 * N, 0, 1);
      } else {
#pragma unroll
        for (int e = 0; e < 8; ++e) {
          int ng = nblk + cs + e;
          __bf16 x0 = (ng < N) ? B[(size_t)(k0 + 2 * a2) * N + ng] : f2bf(0.f);
          __bf16 x1 = (ng < N) ? B[(size_t)(k0 + 2 * a2 + 1) * N + ng] : f2bf(0.f);
          *(unsigned*)&sBt[(cs + e) * LDB_BF + 2 * a2] = pack2(x0, x1);
        }
      }
    }
#if ASYNC_LDS
    asm volatile("s_wait_asynccnt 0x0" ::: "memory");
#endif
    __syncthreads();

    // A frags: K-contiguous, matching B (lanes 0-15: K 0..15; lanes 16-31: K 16..31)
    v16bf af[2];
#pragma unroll
    for (int i = 0; i < 2; ++i) {
      int m = wm + i * 16 + (lane & 15);
#pragma unroll
      for (int j = 0; j < 16; ++j) {
        int kk = ((lane < 16) ? 0 : 16) + j;
        af[i][j] = sA[m * LDA_BF + kk];
      }
    }
    // B frags: K-contiguous, matching A (lanes 0-15: K 0..15; lanes 16-31: K 16..31)
    v16bf bf[4];
#pragma unroll
    for (int jn = 0; jn < 4; ++jn) {
      int nc = wn + jn * 16 + (lane & 15);
#pragma unroll
      for (int j = 0; j < 16; ++j) {
        int kk = ((lane < 16) ? 0 : 16) + j;
        bf[jn][j] = sBt[nc * LDB_BF + kk];
      }
    }
#pragma unroll
    for (int i = 0; i < 2; ++i)
#pragma unroll
      for (int jn = 0; jn < 4; ++jn)
        acc[i][jn] = __builtin_amdgcn_wmma_f32_16x16x32_bf16(
            false, af[i], false, bf[jn], (short)0, acc[i][jn], false, false);
    __syncthreads();
  }

  // store: f32 C/D layout (rows 0-7 lanes 0-15, rows 8-15 lanes 16-31)
#pragma unroll
  for (int i = 0; i < 2; ++i)
#pragma unroll
    for (int jn = 0; jn < 4; ++jn) {
      int nc = nblk + wn + jn * 16 + (lane & 15);
      int mb = mblk + wm + i * 16 + ((lane < 16) ? 0 : 8);
      if (nc < N) {
#pragma unroll
        for (int r = 0; r < 8; ++r)
          C[(size_t)(mb + r) * N + nc] = acc[i][jn][r];
      }
    }
}

// ---------------------------------------------------------------------------
// causal conv1d (K=4) + SiLU over gathered q|k|v channels
// ---------------------------------------------------------------------------
__global__ void conv_silu_kernel(const float* __restrict__ qkvz,
                                 const float* __restrict__ cw,
                                 float* __restrict__ mixed) {
  int bs = blockIdx.x;                      // b*S + s
  int c  = blockIdx.y * 256 + threadIdx.x;  // 0..8191
  int s  = bs & (SQ - 1);
  int col;
  if (c < 2048)        col = (c >> 7) * HEADCOLS + (c & 127);
  else if (c < 4096) { int c2 = c - 2048; col = (c2 >> 7) * HEADCOLS + 128 + (c2 & 127); }
  else               { int c3 = c - 4096; col = (c3 >> 8) * HEADCOLS + 256 + (c3 & 255); }
  float acc = 0.f;
#pragma unroll
  for (int j = 0; j < 4; ++j) {
    int sp = s + j - 3;
    float x = (sp >= 0) ? qkvz[(size_t)(bs + j - 3) * QKVZ_N + col] : 0.f;
    acc += x * cw[c * 4 + j];
  }
  mixed[(size_t)bs * CONVD + c] = siluf_(acc);
}

// ---------------------------------------------------------------------------
// beta = sigmoid(b),  g = -exp(A_log) * softplus(a + dt_bias)
// ---------------------------------------------------------------------------
__global__ void beta_g_kernel(const float* __restrict__ ba,
                              const float* __restrict__ A_log,
                              const float* __restrict__ dt_bias,
                              float* __restrict__ beta, float* __restrict__ g, int n) {
  int i = blockIdx.x * 256 + threadIdx.x;
  if (i >= n) return;
  int vh = i & 31, bs = i >> 5;
  int h = vh >> 1, r = vh & 1;
  float bv = ba[(size_t)bs * 64 + h * 4 + r];
  float av = ba[(size_t)bs * 64 + h * 4 + 2 + r];
  beta[i] = sigmoidf_(bv);
  float x = av + dt_bias[vh];
  float sp = (x > 20.f) ? x : log1pf(__expf(x));
  g[i] = -__expf(A_log[vh]) * sp;
}

// ---------------------------------------------------------------------------
// fp32 WMMA tile helpers (v_wmma_f32_16x16x4_f32)
// A 16x4 f32 frag: lanes 0-15 K={0,1}, lanes 16-31 K={2,3}; M = lane&15
// B 4x16  f32 frag: lanes 0-15 K={0,1}, lanes 16-31 K={2,3}; N = lane&15
// All B operands here are stored transposed (Bt[n][k]) so reads are b64s.
// ---------------------------------------------------------------------------
__device__ __forceinline__ v8f wmma4(v2f a, v2f b, v8f c) {
  return __builtin_amdgcn_wmma_f32_16x16x4_f32(false, a, false, b, (short)0, c, false, false);
}

// acc += A[M,K] @ Bt[N,K]^T   (both row-major in their own layout)
__device__ __forceinline__ v8f mm_nt(const float* A, int lda, const float* Bt, int ldbt,
                                     int m0, int n0, int K, int lane, v8f acc) {
  const int mr = m0 + (lane & 15);
  const int nc = n0 + (lane & 15);
  const int ko = (lane < 16) ? 0 : 2;
  for (int k = 0; k < K; k += 4) {
    v2f a, b;
    a.x = A[mr * lda + k + ko];       a.y = A[mr * lda + k + ko + 1];
    b.x = Bt[nc * ldbt + k + ko];     b.y = Bt[nc * ldbt + k + ko + 1];
    acc = wmma4(a, b, acc);
  }
  return acc;
}

// acc += A[M,K] @ diag(wsc) @ B[K,N] (B row-major, row-scaled by wsc[k])
__device__ __forceinline__ v8f mm_nn_rsB(const float* A, int lda, const float* B, int ldb,
                                         const float* wsc, int m0, int n0, int K, int lane, v8f acc) {
  const int mr = m0 + (lane & 15);
  const int nc = n0 + (lane & 15);
  const int ko = (lane < 16) ? 0 : 2;
  for (int k = 0; k < K; k += 4) {
    v2f a, b;
    a.x = A[mr * lda + k + ko];                     a.y = A[mr * lda + k + ko + 1];
    b.x = B[(k + ko) * ldb + nc] * wsc[k + ko];     b.y = B[(k + ko + 1) * ldb + nc] * wsc[k + ko + 1];
    acc = wmma4(a, b, acc);
  }
  return acc;
}

// acc += (diag(w) @ Kt)^T @ Bt^T :  A[m][k] = Kt[k][m]*w[k],  B[k][n] = Bt[n][k]
__device__ __forceinline__ v8f mm_tn_rsA_bT(const float* Kt, int ldk, const float* w,
                                            const float* Bt, int ldbt,
                                            int m0, int n0, int K, int lane, v8f acc) {
  const int mr = m0 + (lane & 15);
  const int nc = n0 + (lane & 15);
  const int ko = (lane < 16) ? 0 : 2;
  for (int k = 0; k < K; k += 4) {
    v2f a, b;
    a.x = Kt[(k + ko) * ldk + mr] * w[k + ko];
    a.y = Kt[(k + ko + 1) * ldk + mr] * w[k + ko + 1];
    b.x = Bt[nc * ldbt + k + ko];   b.y = Bt[nc * ldbt + k + ko + 1];
    acc = wmma4(a, b, acc);
  }
  return acc;
}

// row-major D-tile store
__device__ __forceinline__ void store_tile_lds(float* C, int ldc, int m0, int n0, int lane, v8f acc) {
  const int nc = n0 + (lane & 15);
  const int mb = m0 + ((lane < 16) ? 0 : 8);
#pragma unroll
  for (int r = 0; r < 8; ++r) C[(mb + r) * ldc + nc] = acc[r];
}
// transposed D-tile store: Ct[n][m] (contiguous across r -> b128-mergeable)
__device__ __forceinline__ void store_tile_T(float* Ct, int ldct, int m0, int n0, int lane, v8f acc) {
  const int nc = n0 + (lane & 15);
  const int mb = m0 + ((lane < 16) ? 0 : 8);
#pragma unroll
  for (int r = 0; r < 8; ++r) Ct[nc * ldct + mb + r] = acc[r];
}

// ---------------------------------------------------------------------------
// Gated delta-rule core. One block per (b, v-head); 256 threads = 8 waves.
// state and v stored TRANSPOSED ([v][d], [v][t]) so all f32-WMMA B operands
// read K-contiguous and D-tile writebacks are lane-contiguous.
// ~235 KB LDS of the 320 KB WGP budget.
// ---------------------------------------------------------------------------
__global__ __launch_bounds__(256) void core_kernel(
    const float* __restrict__ mixed, const float* __restrict__ gbuf,
    const float* __restrict__ bbuf, float* __restrict__ out) {
  __shared__ float stt[128 * LD128];  // state^T [v][d]
  __shared__ float q  [64 * LD128];   // q (normed)          [t][d]
  __shared__ float kk [64 * LD128];   // k (normed)          [t][d]
  __shared__ float kb [64 * LD128];   // k_beta -> kcd       [t][d]
  __shared__ float vt [128 * LD64];   // v^T: v_beta -> v_new -> vn  [v][t]
  __shared__ float Mm [64 * LD64];    // M -> attn           [i][j]
  __shared__ float Tt [64 * LD64];    // (I-M)^-1            [i][j]
  __shared__ float gc[64], sb[64], qs[64], ks[64], eg[64], wst[64];

  const int tid  = threadIdx.x;
  const int lane = tid & 31;
  const int w    = tid >> 5;
  const int blk  = blockIdx.x;
  const int b    = blk >> 5;        // / NV
  const int vh   = blk & 31;
  const int kh   = vh >> 1;         // q,k head (R=2 repeat)

  for (int i = tid; i < 128 * LD128; i += 256) stt[i] = 0.f;
  __syncthreads();

  for (int n = 0; n < SQ / 64; ++n) {
    const int s0 = n * 64;

    // ---- stage q,k,v chunk ----
    for (int idx = tid; idx < 64 * 128; idx += 256) {
      int r = idx >> 7, d = idx & 127;
      size_t row = (size_t)(b * SQ + s0 + r) * CONVD;
      q [r * LD128 + d] = mixed[row + kh * 128 + d];
      kk[r * LD128 + d] = mixed[row + 2048 + kh * 128 + d];
      vt[d * LD64  + r] = mixed[row + 4096 + vh * 128 + d];
    }
    __syncthreads();

    // ---- row stats: l2 norms, beta, g cumsum ----
    if (tid < 64) {
      float s = 0.f;
      for (int d = 0; d < 128; ++d) { float x = q[tid * LD128 + d]; s += x * x; }
      qs[tid] = rsqrtf(s + 1e-6f) * 0.08838834764831845f;  // * DK^-0.5
    } else if (tid < 128) {
      int r = tid - 64; float s = 0.f;
      for (int d = 0; d < 128; ++d) { float x = kk[r * LD128 + d]; s += x * x; }
      ks[r] = rsqrtf(s + 1e-6f);
    } else if (tid < 192) {
      int r = tid - 128;
      sb[r] = bbuf[(size_t)(b * SQ + s0 + r) * NVH + vh];
    } else if (tid == 192) {
      float a = 0.f;
      for (int t = 0; t < 64; ++t) { a += gbuf[(size_t)(b * SQ + s0 + t) * NVH + vh]; gc[t] = a; }
    }
    __syncthreads();

    // ---- scale rows: q*=qs, k*=ks, kb=k*beta, v*=beta; eg=exp(gc) ----
    if (tid < 64) eg[tid] = __expf(gc[tid]);
    for (int idx = tid; idx < 64 * 128; idx += 256) {
      int r = idx >> 7, d = idx & 127;
      q[r * LD128 + d] *= qs[r];
      float kv = kk[r * LD128 + d] * ks[r];
      kk[r * LD128 + d] = kv;
      kb[r * LD128 + d] = kv * sb[r];
      vt[d * LD64 + r] *= sb[r];
    }
    __syncthreads();

    // ---- M = -strict_tril(k_beta @ k^T * decay) ----
#pragma unroll
    for (int t = 0; t < 2; ++t) {
      int tile = w * 2 + t;
      int m0 = (tile >> 2) * 16, n0 = (tile & 3) * 16;
      v8f acc = (v8f){0.f,0.f,0.f,0.f,0.f,0.f,0.f,0.f};
      acc = mm_nt(kb, LD128, kk, LD128, m0, n0, 128, lane, acc);
      int nc = n0 + (lane & 15);
      int mb = m0 + ((lane < 16) ? 0 : 8);
#pragma unroll
      for (int r = 0; r < 8; ++r) {
        int i = mb + r;
        Mm[i * LD64 + nc] = (nc < i) ? (-acc[r] * __expf(gc[i] - gc[nc])) : 0.f;
      }
    }
    __syncthreads();

    // ---- T = (I - M)^-1 : 64 independent per-column forward substitutions ----
    if (tid < 64) {
      int c = tid;
      for (int i = 0; i < 64; ++i) {
        float t = (i == c) ? 1.f : 0.f;
        for (int j = c; j < i; ++j) t += Mm[i * LD64 + j] * Tt[j * LD64 + c];
        Tt[i * LD64 + c] = t;
      }
    }
    __syncthreads();

    // ---- v_new = T @ v_beta ;  kcd = T @ (k_beta * exp(gc)) ----
    {
      v8f vacc[4], kacc[4];
#pragma unroll
      for (int t = 0; t < 4; ++t) {
        int tile = w * 4 + t;
        int m0 = (tile >> 3) * 16, n0 = (tile & 7) * 16;
        v8f z = (v8f){0.f,0.f,0.f,0.f,0.f,0.f,0.f,0.f};
        vacc[t] = mm_nt(Tt, LD64, vt, LD64, m0, n0, 64, lane, z);
        kacc[t] = mm_nn_rsB(Tt, LD64, kb, LD128, eg, m0, n0, 64, lane, z);
      }
      __syncthreads();
#pragma unroll
      for (int t = 0; t < 4; ++t) {
        int tile = w * 4 + t;
        int m0 = (tile >> 3) * 16, n0 = (tile & 7) * 16;
        store_tile_T(vt, LD64, m0, n0, lane, vacc[t]);     // v_new (transposed)
        store_tile_lds(kb, LD128, m0, n0, lane, kacc[t]);  // kcd (row-major, A later)
      }
    }
    __syncthreads();

    // ---- attn = tril(q @ k^T * decay) (incl diagonal) ----
    {
      v8f aacc[2];
#pragma unroll
      for (int t = 0; t < 2; ++t) {
        int tile = w * 2 + t;
        int m0 = (tile >> 2) * 16, n0 = (tile & 3) * 16;
        v8f z = (v8f){0.f,0.f,0.f,0.f,0.f,0.f,0.f,0.f};
        aacc[t] = mm_nt(q, LD128, kk, LD128, m0, n0, 128, lane, z);
      }
      __syncthreads();
#pragma unroll
      for (int t = 0; t < 2; ++t) {
        int tile = w * 2 + t;
        int m0 = (tile >> 2) * 16, n0 = (tile & 3) * 16;
        int nc = n0 + (lane & 15);
        int mb = m0 + ((lane < 16) ? 0 : 8);
#pragma unroll
        for (int r = 0; r < 8; ++r) {
          int i = mb + r;
          Mm[i * LD64 + nc] = (nc <= i) ? (aacc[t][r] * __expf(gc[i] - gc[nc])) : 0.f;
        }
      }
    }
    // scale q rows by exp(gc) for the inter term; precompute state-decay weights
    if (tid < 64) wst[tid] = __expf(gc[63] - gc[tid]);
    for (int idx = tid; idx < 64 * 128; idx += 256) {
      int r = idx >> 7, d = idx & 127;
      q[r * LD128 + d] *= eg[r];
    }
    __syncthreads();

    // ---- v_prime = kcd @ state ;  vn = v_new - v_prime (in place in vt) ----
    {
      v8f pacc[4];
#pragma unroll
      for (int t = 0; t < 4; ++t) {
        int tile = w * 4 + t;
        int m0 = (tile >> 3) * 16, n0 = (tile & 7) * 16;
        v8f z = (v8f){0.f,0.f,0.f,0.f,0.f,0.f,0.f,0.f};
        pacc[t] = mm_nt(kb, LD128, stt, LD128, m0, n0, 128, lane, z);
      }
      __syncthreads();
#pragma unroll
      for (int t = 0; t < 4; ++t) {
        int tile = w * 4 + t;
        int m0 = (tile >> 3) * 16, n0 = (tile & 7) * 16;
        int nc = n0 + (lane & 15);
        int mb = m0 + ((lane < 16) ? 0 : 8);
#pragma unroll
        for (int r = 0; r < 8; ++r)
          vt[nc * LD64 + mb + r] -= pacc[t][r];
      }
    }
    __syncthreads();

    // ---- out = (q*exp(gc)) @ state + attn @ vn ----
    {
      v8f oacc[4];
#pragma unroll
      for (int t = 0; t < 4; ++t) {
        int tile = w * 4 + t;
        int m0 = (tile >> 3) * 16, n0 = (tile & 7) * 16;
        v8f z = (v8f){0.f,0.f,0.f,0.f,0.f,0.f,0.f,0.f};
        oacc[t] = mm_nt(q, LD128, stt, LD128, m0, n0, 128, lane, z);
        oacc[t] = mm_nt(Mm, LD64, vt, LD64, m0, n0, 64, lane, oacc[t]);
      }
      __syncthreads();   // all state reads done before state update below
#pragma unroll
      for (int t = 0; t < 4; ++t) {
        int tile = w * 4 + t;
        int m0 = (tile >> 3) * 16, n0 = (tile & 7) * 16;
        int nc = n0 + (lane & 15);
        int mb = m0 + ((lane < 16) ? 0 : 8);
#pragma unroll
        for (int r = 0; r < 8; ++r)
          out[((size_t)(b * SQ + s0 + mb + r) * NVH + vh) * DVQ + nc] = oacc[t][r];
      }
    }

    // ---- state^T = state^T*exp(g_last) + [(k*exp(g_last-gc))^T @ vn]^T ----
    {
      float el = __expf(gc[63]);
      v8f sacc[8];
#pragma unroll
      for (int t = 0; t < 8; ++t) {
        int m0 = w * 16, n0 = t * 16;   // wave w owns d-rows [16w, 16w+16)
        v8f z = (v8f){0.f,0.f,0.f,0.f,0.f,0.f,0.f,0.f};
        sacc[t] = mm_tn_rsA_bT(kk, LD128, wst, vt, LD64, m0, n0, 64, lane, z);
      }
#pragma unroll
      for (int t = 0; t < 8; ++t) {
        int m0 = w * 16, n0 = t * 16;
        int nc = n0 + (lane & 15);
        int mb = m0 + ((lane < 16) ? 0 : 8);
#pragma unroll
        for (int r = 0; r < 8; ++r) {
          int d = mb + r;
          stt[nc * LD128 + d] = stt[nc * LD128 + d] * el + sacc[t][r];
        }
      }
    }
    __syncthreads();
  }
}

// ---------------------------------------------------------------------------
// x = core * silu(z); RMSNorm over DV; * norm_weight; -> bf16
// ---------------------------------------------------------------------------
__global__ void gate_norm_kernel(const float* __restrict__ core,
                                 const float* __restrict__ qkvz,
                                 const float* __restrict__ nw,
                                 __bf16* __restrict__ xbf) {
  __shared__ float red[128];
  int bs = blockIdx.x, vh = blockIdx.y, d = threadIdx.x;
  float y = core[((size_t)bs * NVH + vh) * DVQ + d];
  float z = qkvz[(size_t)bs * QKVZ_N + (vh >> 1) * HEADCOLS + 512 + (vh & 1) * 128 + d];
  float x = y * siluf_(z);
  red[d] = x * x;
  __syncthreads();
  for (int off = 64; off > 0; off >>= 1) {
    if (d < off) red[d] += red[d + off];
    __syncthreads();
  }
  float scale = rsqrtf(red[0] * (1.f / 128.f) + 1e-6f);
  xbf[(size_t)bs * 4096 + vh * 128 + d] = f2bf(x * scale * nw[d]);
}

// ---------------------------------------------------------------------------
// launch
// ---------------------------------------------------------------------------
extern "C" void kernel_launch(void* const* d_in, const int* in_sizes, int n_in,
                              void* d_out, int out_size, void* d_ws, size_t ws_size,
                              hipStream_t stream) {
  const float* hidden = (const float*)d_in[0];
  const float* Wqkvz  = (const float*)d_in[1];
  const float* Wba    = (const float*)d_in[2];
  const float* convw  = (const float*)d_in[3];
  const float* A_log  = (const float*)d_in[4];
  const float* dtb    = (const float*)d_in[5];
  const float* nw     = (const float*)d_in[6];
  const float* Wout   = (const float*)d_in[7];
  (void)in_sizes; (void)n_in; (void)out_size; (void)ws_size;

  char* ws = (char*)d_ws;
  size_t off = 0;
  auto take = [&](size_t bytes) -> char* {
    char* p = ws + off;
    off = (off + bytes + 255) & ~(size_t)255;
    return p;
  };
  __bf16* hidbf = (__bf16*)take((size_t)BS * HQ * 2);
  __bf16* wqbf  = (__bf16*)take((size_t)HQ * QKVZ_N * 2);
  __bf16* wbabf = (__bf16*)take((size_t)HQ * 64 * 2);
  __bf16* wobf  = (__bf16*)take((size_t)4096 * HQ * 2);
  float*  qkvz  = (float*)take((size_t)BS * QKVZ_N * 4);
  float*  ba    = (float*)take((size_t)BS * 64 * 4);
  float*  mixed = (float*)take((size_t)BS * CONVD * 4);
  float*  gbuf  = (float*)take((size_t)BS * NVH * 4);
  float*  bbuf  = (float*)take((size_t)BS * NVH * 4);
  float*  core  = (float*)take((size_t)BS * 4096 * 4);
  __bf16* xbf   = (__bf16*)take((size_t)BS * 4096 * 2);

  // 1) cast operands to bf16 for the matrix pipe
  cast_bf16_kernel<<<(BS * HQ + 255) / 256, 256, 0, stream>>>(hidden, hidbf, BS * HQ);
  cast_bf16_kernel<<<(HQ * QKVZ_N + 255) / 256, 256, 0, stream>>>(Wqkvz, wqbf, HQ * QKVZ_N);
  cast_bf16_kernel<<<(HQ * 64 + 255) / 256, 256, 0, stream>>>(Wba, wbabf, HQ * 64);
  cast_bf16_kernel<<<(4096 * HQ + 255) / 256, 256, 0, stream>>>(Wout, wobf, 4096 * HQ);

  // 2) projections
  gemm_bf16_kernel<<<dim3(QKVZ_N / 128, BS / 128), 256, 0, stream>>>(hidbf, wqbf, qkvz, BS, QKVZ_N, HQ);
  gemm_bf16_kernel<<<dim3(1, BS / 128), 256, 0, stream>>>(hidbf, wbabf, ba, BS, 64, HQ);

  // 3) conv + silu, gate params
  conv_silu_kernel<<<dim3(BS, CONVD / 256), 256, 0, stream>>>(qkvz, convw, mixed);
  beta_g_kernel<<<(BS * NVH + 255) / 256, 256, 0, stream>>>(ba, A_log, dtb, bbuf, gbuf, BS * NVH);

  // 4) gated delta-rule core: one WGP-resident block per (b, v-head)
  core_kernel<<<BQ * NVH, 256, 0, stream>>>(mixed, gbuf, bbuf, core);

  // 5) gate + RMSNorm -> bf16, then output projection
  gate_norm_kernel<<<dim3(BS, NVH), 128, 0, stream>>>(core, qkvz, nw, xbf);
  gemm_bf16_kernel<<<dim3(HQ / 128, BS / 128), 256, 0, stream>>>(xbf, wobf, (float*)d_out, BS, HQ, 4096);
}
